// TechBookGCN_18674517803652
// MI455X (gfx1250) — matrix operand order repaired and checked by reading, not compiled, not run on hardware
//
#include <hip/hip_runtime.h>

// ---------------------------------------------------------------------------
// 3-layer GCN (GCNConv -> BN -> ReLU) x2 -> GCNConv -> log_softmax
// MI455X / gfx1250: bf16 WMMA GEMMs (v_wmma_f32_16x16x32_bf16),
// async global->LDS copies (ASYNCcnt), LDS double-buffering, f32 elsewhere.
// ---------------------------------------------------------------------------

typedef __attribute__((ext_vector_type(16))) __bf16 v16bf;
typedef __attribute__((ext_vector_type(8)))  float  v8f;

#define GCN_EPS 1e-5f

// ---- bf16 <-> f32 bit helpers (storage-only use of __bf16) ----------------
__device__ __forceinline__ __bf16 f2bf(float f) {
    union { float f; unsigned u; } in; in.f = f;
    unsigned u = in.u;
    u += 0x7FFFu + ((u >> 16) & 1u);          // round-to-nearest-even
    union { unsigned short s; __bf16 b; } out;
    out.s = (unsigned short)(u >> 16);
    return out.b;
}
__device__ __forceinline__ __bf16 bf_zero() {
    union { unsigned short s; __bf16 b; } out; out.s = 0; return out.b;
}

// ---------------------------------------------------------------------------
// Elementwise helpers
// ---------------------------------------------------------------------------
__global__ void cvt_f32_bf16(const float* __restrict__ src, __bf16* __restrict__ dst,
                             long long n) {
    long long i = (long long)blockIdx.x * blockDim.x + threadIdx.x;
    if (i < n) dst[i] = f2bf(src[i]);
}

// pad W3 (K x 5, f32) -> (K x 16, bf16), zero cols 5..15
__global__ void pad_w3_kernel(const float* __restrict__ w, __bf16* __restrict__ dst,
                              int K) {
    int i = blockIdx.x * blockDim.x + threadIdx.x;
    if (i >= K * 16) return;
    int k = i >> 4, j = i & 15;
    dst[i] = (j < 5) ? f2bf(w[k * 5 + j]) : bf_zero();
}

__global__ void deg_init_kernel(float* __restrict__ deg, int n) {
    int i = blockIdx.x * blockDim.x + threadIdx.x;
    if (i < n) deg[i] = 1.0f;                 // self loop contributes 1
}

__global__ void deg_accum_kernel(const int* __restrict__ dst_idx,
                                 float* __restrict__ deg, int E) {
    int e = blockIdx.x * blockDim.x + threadIdx.x;
    if (e < E) atomicAdd(&deg[dst_idx[e]], 1.0f);
}

__global__ void isq_kernel(const float* __restrict__ deg, float* __restrict__ isq,
                           int n) {
    int i = blockIdx.x * blockDim.x + threadIdx.x;
    if (i < n) isq[i] = rsqrtf(deg[i]);       // deg >= 1 always (self loops)
}

__global__ void norm_kernel(const int* __restrict__ src, const int* __restrict__ dst,
                            const float* __restrict__ isq, float* __restrict__ nrm,
                            int E) {
    int e = blockIdx.x * blockDim.x + threadIdx.x;
    if (e < E) nrm[e] = isq[src[e]] * isq[dst[e]];
}

// ---------------------------------------------------------------------------
// GEMM: C[M,Nc] = A[M,K] (bf16, row-major) * B[K,Nc] (bf16, row-major), C f32.
// Block tile 64x128x32, 8 waves, wave tile 16x64 (4 WMMA accumulators).
// Double-buffered LDS; A tile via async global->LDS b128 (ASYNCcnt);
// B tile transposed into LDS so both fragment types are K-contiguous b128s.
// ---------------------------------------------------------------------------
#define BM 64
#define BN_T 128
#define BK 32
#define LDA  40    // bf16 elems: 80B row stride (16B aligned, conflict-free)
#define LDBT 40    // bf16 elems per B column in LDS (32 K + 8 pad), 80B stride

__global__ __launch_bounds__(256)
void gemm_bf16_wmma(const __bf16* __restrict__ A, const __bf16* __restrict__ B,
                    float* __restrict__ C, int M, int Nc, int K) {
    __shared__ __bf16 sA[2][BM * LDA];        // [buf][row][k] row-major
    __shared__ __bf16 sBT[2][BN_T * LDBT];    // [buf][n][k]  column-major

    const int tid  = threadIdx.x;
    const int wave = tid >> 5;
    const int lane = tid & 31;
    const int g    = lane >> 4;       // lane group (0/1)
    const int ln   = lane & 15;

    const int block_m = blockIdx.y * BM;
    const int block_n = blockIdx.x * BN_T;

    const int wm = (wave & 3) * 16;   // 4 waves in M
    const int wn = (wave >> 2) * 64;  // 2 waves in N

    // ---- per-thread tile-copy coordinates (loop-invariant) ----
    const int a_row = tid >> 2;              // 0..63
    const int a_col = (tid & 3) * 8;         // 0,8,16,24
    int gm = block_m + a_row;
    if (gm >= M) gm = M - 1;                 // clamp: rows >= M feed unstored outputs
    const unsigned a_lds0 = (unsigned)(unsigned long long)(sA[0] + a_row * LDA + a_col);
    const unsigned a_lds1 = (unsigned)(unsigned long long)(sA[1] + a_row * LDA + a_col);
    unsigned long long ga = (unsigned long long)(const void*)(A + (size_t)gm * K + a_col);

    const int b_row = tid >> 3;              // 0..31  (k within tile)
    const int b_col = (tid & 7) * 16;        // 0..112 (n within tile)
    const bool bg0 = (block_n + b_col + 8  <= Nc);
    const bool bg1 = (block_n + b_col + 16 <= Nc);
    const __bf16* bp = B + (size_t)b_row * Nc + block_n + b_col;
    const size_t bstep = (size_t)BK * Nc;

    // ---- one-time zero-fill of OOB B columns (both buffers) ----
    if (!bg0) {
        #pragma unroll
        for (int j = 0; j < 8; ++j) {
            sBT[0][(size_t)(b_col + j) * LDBT + b_row] = bf_zero();
            sBT[1][(size_t)(b_col + j) * LDBT + b_row] = bf_zero();
        }
    }
    if (!bg1) {
        #pragma unroll
        for (int j = 0; j < 8; ++j) {
            sBT[0][(size_t)(b_col + 8 + j) * LDBT + b_row] = bf_zero();
            sBT[1][(size_t)(b_col + 8 + j) * LDBT + b_row] = bf_zero();
        }
    }

    union { uint4 v; __bf16 h[8]; } u0, u1;

    // ---- prologue: stage tile 0 into buffer 0 ----
    asm volatile("global_load_async_to_lds_b128 %0, %1, off"
                 :: "v"(a_lds0), "v"(ga) : "memory");
    ga += (unsigned long long)BK * 2;
    if (bg0) u0.v = *reinterpret_cast<const uint4*>(bp);
    if (bg1) u1.v = *reinterpret_cast<const uint4*>(bp + 8);
    bp += bstep;
    {
        __bf16* outc = sBT[0] + (size_t)b_col * LDBT + b_row;
        if (bg0) {
            #pragma unroll
            for (int j = 0; j < 8; ++j) outc[(size_t)j * LDBT] = u0.h[j];
        }
        if (bg1) {
            #pragma unroll
            for (int j = 0; j < 8; ++j) outc[(size_t)(8 + j) * LDBT] = u1.h[j];
        }
    }
    asm volatile("s_wait_asynccnt 0" ::: "memory");
    __syncthreads();

    v8f acc[4] = {};
    const int nt = K / BK;

    for (int i = 0; i < nt; ++i) {
        const int cur = i & 1;
        const int nxt = cur ^ 1;
        const bool more = (i + 1 < nt);

        // --- prefetch tile i+1: async A copy + B global loads (in flight over compute)
        if (more) {
            unsigned a_dst = nxt ? a_lds1 : a_lds0;
            asm volatile("global_load_async_to_lds_b128 %0, %1, off"
                         :: "v"(a_dst), "v"(ga) : "memory");
            ga += (unsigned long long)BK * 2;
            if (bg0) u0.v = *reinterpret_cast<const uint4*>(bp);
            if (bg1) u1.v = *reinterpret_cast<const uint4*>(bp + 8);
            bp += bstep;
        }

        // --- compute tile i from buffer cur
        const __bf16* sAc = sA[cur];
        const __bf16* sBc = sBT[cur];
        v16bf af;
        {
            const __bf16* ap = sAc + (wm + ln) * LDA;
            #pragma unroll
            for (int r = 0; r < 8; ++r) af[r]     = ap[g * 16 + r];       // K=g*16+0..7
            #pragma unroll
            for (int r = 0; r < 8; ++r) af[8 + r] = ap[g * 16 + 8 + r];   // K=g*16+8..15
        }
        #pragma unroll
        for (int nb = 0; nb < 4; ++nb) {
            v16bf bfrag;
            const __bf16* bfp = sBc + (size_t)(wn + nb * 16 + ln) * LDBT + g * 16;
            #pragma unroll
            for (int r = 0; r < 16; ++r) bfrag[r] = bfp[r];               // K=g*16+0..15
            acc[nb] = __builtin_amdgcn_wmma_f32_16x16x32_bf16(
                false, af, false, bfrag, (short)0, acc[nb], false, false);
        }

        // --- commit tile i+1 B regs to LDS, wait async, single barrier
        if (more) {
            __bf16* outc = sBT[nxt] + (size_t)b_col * LDBT + b_row;
            if (bg0) {
                #pragma unroll
                for (int j = 0; j < 8; ++j) outc[(size_t)j * LDBT] = u0.h[j];
            }
            if (bg1) {
                #pragma unroll
                for (int j = 0; j < 8; ++j) outc[(size_t)(8 + j) * LDBT] = u1.h[j];
            }
            asm volatile("s_wait_asynccnt 0" ::: "memory");
            __syncthreads();
        }
    }

    // --- store (C/D layout: m = r + 8*g across VGPRs, n across lanes)
    #pragma unroll
    for (int nb = 0; nb < 4; ++nb) {
        int n = block_n + wn + nb * 16 + ln;
        if (n >= Nc) continue;
        #pragma unroll
        for (int r = 0; r < 8; ++r) {
            int m = block_m + wm + g * 8 + r;
            if (m < M) C[(size_t)m * Nc + n] = acc[nb][r];
        }
    }
}

// ---------------------------------------------------------------------------
// Aggregation: AGG[i] = (1/deg_i)*H[i] + bias  (self loop), then edge scatter
// ---------------------------------------------------------------------------
__global__ void agg_init_kernel(const float* __restrict__ H,
                                const float* __restrict__ isq,
                                const float* __restrict__ bias,
                                float* __restrict__ AGG,
                                int Nn, int d, int ldh) {
    long long idx = (long long)blockIdx.x * blockDim.x + threadIdx.x;
    long long total = (long long)Nn * d;
    if (idx >= total) return;
    int i = (int)(idx / d);
    int c = (int)(idx - (long long)i * d);
    float s = isq[i];
    AGG[(size_t)i * ldh + c] = H[(size_t)i * ldh + c] * (s * s) + bias[c];
}

__global__ void agg_edges_vec4(const int* __restrict__ src, const int* __restrict__ dst,
                               const float* __restrict__ nrm,
                               const float* __restrict__ H, float* __restrict__ AGG,
                               int E, int dq, int d) {
    long long idx = (long long)blockIdx.x * blockDim.x + threadIdx.x;
    long long total = (long long)E * dq;
    if (idx >= total) return;
    int e = (int)(idx / dq);
    int q = (int)(idx - (long long)e * dq);
    int s = src[e], t = dst[e];
    float w = nrm[e];
    const float4 hv = *reinterpret_cast<const float4*>(H + (size_t)s * d + q * 4);
    float* out = AGG + (size_t)t * d + q * 4;
    atomicAdd(out + 0, w * hv.x);
    atomicAdd(out + 1, w * hv.y);
    atomicAdd(out + 2, w * hv.z);
    atomicAdd(out + 3, w * hv.w);
}

// layer 3: 5 features, stride 16
__global__ void agg_edges_d5(const int* __restrict__ src, const int* __restrict__ dst,
                             const float* __restrict__ nrm,
                             const float* __restrict__ H, float* __restrict__ AGG,
                             int E) {
    int e = blockIdx.x * blockDim.x + threadIdx.x;
    if (e >= E) return;
    int s = src[e], t = dst[e];
    float w = nrm[e];
    #pragma unroll
    for (int j = 0; j < 5; ++j)
        atomicAdd(AGG + (size_t)t * 16 + j, w * H[(size_t)s * 16 + j]);
}

// ---------------------------------------------------------------------------
// BatchNorm (training-mode batch stats) + ReLU + bf16 convert
// ---------------------------------------------------------------------------
__global__ void bn_zero_kernel(float* __restrict__ sums, float* __restrict__ sumsq,
                               int d) {
    int c = blockIdx.x * blockDim.x + threadIdx.x;
    if (c < d) { sums[c] = 0.0f; sumsq[c] = 0.0f; }
}

#define BN_ROWS 128
__global__ void bn_stats_kernel(const float* __restrict__ A, float* __restrict__ sums,
                                float* __restrict__ sumsq, int Nn, int d) {
    int r0 = blockIdx.x * BN_ROWS;
    int r1 = r0 + BN_ROWS; if (r1 > Nn) r1 = Nn;
    for (int c = threadIdx.x; c < d; c += blockDim.x) {
        float s = 0.0f, q = 0.0f;
        for (int r = r0; r < r1; ++r) {
            float v = A[(size_t)r * d + c];
            s += v; q += v * v;
        }
        atomicAdd(sums + c, s);
        atomicAdd(sumsq + c, q);
    }
}

__global__ void bn_finalize_kernel(const float* __restrict__ sums,
                                   const float* __restrict__ sumsq,
                                   const float* __restrict__ gamma,
                                   const float* __restrict__ beta,
                                   float* __restrict__ scale,
                                   float* __restrict__ shift,
                                   int d, float invN) {
    int c = blockIdx.x * blockDim.x + threadIdx.x;
    if (c >= d) return;
    float mean = sums[c] * invN;
    float var  = sumsq[c] * invN - mean * mean;
    float sc   = gamma[c] * rsqrtf(var + GCN_EPS);
    scale[c] = sc;
    shift[c] = beta[c] - mean * sc;
}

__global__ void bn_act_bf16_kernel(const float* __restrict__ A,
                                   const float* __restrict__ scale,
                                   const float* __restrict__ shift,
                                   __bf16* __restrict__ out,
                                   long long n, int d) {
    long long idx = (long long)blockIdx.x * blockDim.x + threadIdx.x;
    if (idx >= n) return;
    int c = (int)(idx % d);
    float v = A[idx] * scale[c] + shift[c];
    out[idx] = f2bf(fmaxf(v, 0.0f));
}

// ---------------------------------------------------------------------------
// log_softmax over 5 classes (input stride 16, output stride 5)
// ---------------------------------------------------------------------------
__global__ void logsoftmax5_kernel(const float* __restrict__ A, float* __restrict__ out,
                                   int Nn) {
    int i = blockIdx.x * blockDim.x + threadIdx.x;
    if (i >= Nn) return;
    float v[5];
    #pragma unroll
    for (int j = 0; j < 5; ++j) v[j] = A[(size_t)i * 16 + j];
    float m = v[0];
    #pragma unroll
    for (int j = 1; j < 5; ++j) m = fmaxf(m, v[j]);
    float se = 0.0f;
    #pragma unroll
    for (int j = 0; j < 5; ++j) se += expf(v[j] - m);
    float lse = m + logf(se);
    #pragma unroll
    for (int j = 0; j < 5; ++j) out[(size_t)i * 5 + j] = v[j] - lse;
}

// ---------------------------------------------------------------------------
// Host: orchestration (all launches on `stream`; graph-capture safe)
// ---------------------------------------------------------------------------
static inline unsigned gblk(long long n, int b) { return (unsigned)((n + b - 1) / b); }

extern "C" void kernel_launch(void* const* d_in, const int* in_sizes, int n_in,
                              void* d_out, int out_size, void* d_ws, size_t ws_size,
                              hipStream_t stream) {
    const int D_IN = 1536, D_H = 512, D_MID = 128;
    const int Nn = in_sizes[0] / D_IN;       // 50000
    const int E  = in_sizes[1] / 2;          // 800000

    const float* x   = (const float*)d_in[0];
    const int*   ei  = (const int*)d_in[1];
    const float* W1  = (const float*)d_in[2];
    const float* b1  = (const float*)d_in[3];
    const float* W2  = (const float*)d_in[4];
    const float* b2  = (const float*)d_in[5];
    const float* W3  = (const float*)d_in[6];
    const float* b3  = (const float*)d_in[7];
    const float* g1  = (const float*)d_in[8];
    const float* be1 = (const float*)d_in[9];
    const float* g2  = (const float*)d_in[10];
    const float* be2 = (const float*)d_in[11];
    float* outp = (float*)d_out;

    const int* e_src = ei;
    const int* e_dst = ei + E;

    // ---- workspace layout (256B aligned) ----
    char* ws = (char*)d_ws;
    size_t off = 0;
    auto take = [&](size_t bytes) -> char* {
        char* p = ws + off;
        off += (bytes + 255) & ~(size_t)255;
        return p;
    };
    __bf16* X16   = (__bf16*)take((size_t)Nn * D_IN * 2);   // 153.6 MB
    __bf16* ACT16 = (__bf16*)take((size_t)Nn * D_H * 2);    // 51.2 MB (reused)
    __bf16* W16   = (__bf16*)take((size_t)D_IN * D_H * 2);  // 1.6 MB (reused)
    float*  H     = (float*)take((size_t)Nn * D_H * 4);     // 102.4 MB (reused)
    float*  AGG   = (float*)take((size_t)Nn * D_H * 4);     // 102.4 MB (reused)
    float*  deg   = (float*)take((size_t)Nn * 4);
    float*  isq   = (float*)take((size_t)Nn * 4);
    float*  nrm   = (float*)take((size_t)E * 4);
    float*  sums  = (float*)take((size_t)D_H * 4);
    float*  sumsq = (float*)take((size_t)D_H * 4);
    float*  scale = (float*)take((size_t)D_H * 4);
    float*  shift = (float*)take((size_t)D_H * 4);
    if (off > ws_size) return;  // workspace too small; nothing safe to do

    const int T = 256;
    dim3 blk(T);

    // ---- degrees / normalization ----
    deg_init_kernel<<<gblk(Nn, T), blk, 0, stream>>>(deg, Nn);
    deg_accum_kernel<<<gblk(E, T), blk, 0, stream>>>(e_dst, deg, E);
    isq_kernel<<<gblk(Nn, T), blk, 0, stream>>>(deg, isq, Nn);
    norm_kernel<<<gblk(E, T), blk, 0, stream>>>(e_src, e_dst, isq, nrm, E);

    // ---- layer 1: x @ W1 -> aggregate -> BN -> ReLU ----
    cvt_f32_bf16<<<gblk((long long)Nn * D_IN, T), blk, 0, stream>>>(x, X16, (long long)Nn * D_IN);
    cvt_f32_bf16<<<gblk((long long)D_IN * D_H, T), blk, 0, stream>>>(W1, W16, (long long)D_IN * D_H);
    {
        dim3 grid((D_H + BN_T - 1) / BN_T, (Nn + BM - 1) / BM);
        gemm_bf16_wmma<<<grid, blk, 0, stream>>>(X16, W16, H, Nn, D_H, D_IN);
    }
    agg_init_kernel<<<gblk((long long)Nn * D_H, T), blk, 0, stream>>>(H, isq, b1, AGG, Nn, D_H, D_H);
    agg_edges_vec4<<<gblk((long long)E * (D_H / 4), T), blk, 0, stream>>>(e_src, e_dst, nrm, H, AGG, E, D_H / 4, D_H);
    bn_zero_kernel<<<gblk(D_H, T), blk, 0, stream>>>(sums, sumsq, D_H);
    bn_stats_kernel<<<gblk((long long)Nn, BN_ROWS), blk, 0, stream>>>(AGG, sums, sumsq, Nn, D_H);
    bn_finalize_kernel<<<gblk(D_H, T), blk, 0, stream>>>(sums, sumsq, g1, be1, scale, shift, D_H, 1.0f / (float)Nn);
    bn_act_bf16_kernel<<<gblk((long long)Nn * D_H, T), blk, 0, stream>>>(AGG, scale, shift, ACT16, (long long)Nn * D_H, D_H);

    // ---- layer 2 ----
    cvt_f32_bf16<<<gblk((long long)D_H * D_MID, T), blk, 0, stream>>>(W2, W16, (long long)D_H * D_MID);
    {
        dim3 grid((D_MID + BN_T - 1) / BN_T, (Nn + BM - 1) / BM);
        gemm_bf16_wmma<<<grid, blk, 0, stream>>>(ACT16, W16, H, Nn, D_MID, D_H);
    }
    agg_init_kernel<<<gblk((long long)Nn * D_MID, T), blk, 0, stream>>>(H, isq, b2, AGG, Nn, D_MID, D_MID);
    agg_edges_vec4<<<gblk((long long)E * (D_MID / 4), T), blk, 0, stream>>>(e_src, e_dst, nrm, H, AGG, E, D_MID / 4, D_MID);
    bn_zero_kernel<<<gblk(D_MID, T), blk, 0, stream>>>(sums, sumsq, D_MID);
    bn_stats_kernel<<<gblk((long long)Nn, BN_ROWS), blk, 0, stream>>>(AGG, sums, sumsq, Nn, D_MID);
    bn_finalize_kernel<<<gblk(D_MID, T), blk, 0, stream>>>(sums, sumsq, g2, be2, scale, shift, D_MID, 1.0f / (float)Nn);
    bn_act_bf16_kernel<<<gblk((long long)Nn * D_MID, T), blk, 0, stream>>>(AGG, scale, shift, ACT16, (long long)Nn * D_MID, D_MID);

    // ---- layer 3: (N,128)@(128,5 padded to 16) -> aggregate -> log_softmax ----
    pad_w3_kernel<<<gblk(D_MID * 16, T), blk, 0, stream>>>(W3, W16, D_MID);
    {
        dim3 grid((16 + BN_T - 1) / BN_T, (Nn + BM - 1) / BM);
        gemm_bf16_wmma<<<grid, blk, 0, stream>>>(ACT16, W16, H, Nn, 16, D_MID);
    }
    agg_init_kernel<<<gblk((long long)Nn * 5, T), blk, 0, stream>>>(H, isq, b3, AGG, Nn, 5, 16);
    agg_edges_d5<<<gblk(E, T), blk, 0, stream>>>(e_src, e_dst, nrm, H, AGG, E);
    logsoftmax5_kernel<<<gblk(Nn, T), blk, 0, stream>>>(AGG, outp, Nn);
    (void)out_size; (void)n_in;
}
